// GraphGenRnn_44246753083823
// MI455X (gfx1250) — matrix operand order repaired and checked by reading, not compiled, not run on hardware
//
#include <hip/hip_runtime.h>
#include <hip/hip_bf16.h>
#include <math.h>

// ============================================================================
// GraphGen RNN for MI455X (gfx1250), wave32 + WMMA bf16.
//
// Strategy:
//  * Precompute gene@Wih_n^T + biases once (constant over node steps).
//  * Node LSTM chain: 62 sequential (cheap) steps via small stream launches.
//  * Precompute E[s] = [gene, entry_s] @ Wih_e^T + biases for ALL node steps
//    as one massive parallel WMMA GEMM (constant over inner edge steps).
//  * 62 independent edge-LSTM chains -> 62 persistent workgroups (32 waves),
//    each running only j = 0..s steps. Gate GEMM (256x2048, K=512) fuses the
//    LSTM cell update into the WMMA epilogue (4 gate tiles per wave-unit).
//  * All GEMMs use v_wmma_f32_16x16x32_bf16 (fp32 accumulate).
// Workspace budget ~200 MB.
// ============================================================================

typedef __attribute__((ext_vector_type(16))) __bf16 v16bf;
typedef __attribute__((ext_vector_type(8)))  float  v8f;
typedef unsigned short u16;

#define WMMA_BF16(a, b, c) \
  __builtin_amdgcn_wmma_f32_16x16x32_bf16(false, (a), false, (b), (short)0, (c), false, false)

#define S_STEPS 62   // N-2 node steps == max edge steps

__device__ __forceinline__ u16 f2bf(float f) {
  unsigned int u = __float_as_uint(f);
  u += 0x7FFFu + ((u >> 16) & 1u);           // round-to-nearest-even
  return (u16)(u >> 16);
}
__device__ __forceinline__ float bf2f(u16 h) {
  return __uint_as_float(((unsigned int)h) << 16);
}
__device__ __forceinline__ float sigm(float x) { return 1.0f / (1.0f + expf(-x)); }

union FragU { v16bf v; unsigned int u[8]; };

// A fragment: 16x32 bf16, A row-major (M x K). ISA 7.12.2:
// lanes 0-15 row M=lane, K={0..7,16..23}; lanes 16-31 row M=lane-16, K={8..15,24..31}.
__device__ __forceinline__ v16bf load_a_frag(const u16* A, int lda, int m0, int k0) {
  int lane = threadIdx.x & 31;
  const u16* row = A + (size_t)(m0 + (lane & 15)) * lda + k0 + ((lane >> 4) << 3);
  FragU r;
#pragma unroll
  for (int v = 0; v < 4; ++v) r.u[v]     = *(const unsigned int*)(row + 2 * v);
#pragma unroll
  for (int v = 0; v < 4; ++v) r.u[4 + v] = *(const unsigned int*)(row + 16 + 2 * v);
  return r.v;
}

// B fragment: 32x16 bf16, B = W^T where W is (N x K) row-major.
// lanes 0-15 col N=lane, K={0..7,16..23}; lanes 16-31 col N=lane-16, K={8..15,24..31}.
__device__ __forceinline__ v16bf load_b_frag(const u16* W, int ldw, int n0, int k0) {
  int lane = threadIdx.x & 31;
  const u16* row = W + (size_t)(n0 + (lane & 15)) * ldw + k0 + ((lane >> 4) << 3);
  FragU r;
#pragma unroll
  for (int v = 0; v < 8; ++v) r.u[v] = *(const unsigned int*)(row + (v >> 2) * 16 + 2 * (v & 3));
  return r.v;
}

// ---------------------------------------------------------------------------
// Elementwise / prep kernels
// ---------------------------------------------------------------------------
__global__ void zero_f32_kernel(float* p, long long n) {
  long long i = (long long)blockIdx.x * blockDim.x + threadIdx.x;
  long long st = (long long)gridDim.x * blockDim.x;
  for (; i < n; i += st) p[i] = 0.0f;
}

__global__ void cvt_bf16_kernel(const float* s, u16* d, int n) {
  int i = blockIdx.x * blockDim.x + threadIdx.x, st = gridDim.x * blockDim.x;
  for (; i < n; i += st) d[i] = f2bf(s[i]);
}

__global__ void copy_f32_kernel(const float* s, float* d, int n) {
  int i = blockIdx.x * blockDim.x + threadIdx.x, st = gridDim.x * blockDim.x;
  for (; i < n; i += st) d[i] = s[i];
}

__global__ void add_f32_kernel(const float* a, const float* b, float* d, int n) {
  int i = blockIdx.x * blockDim.x + threadIdx.x, st = gridDim.x * blockDim.x;
  for (; i < n; i += st) d[i] = a[i] + b[i];
}

// Zero-pad Ws2 (2x128) -> 16x128 bf16 and bs2 (2) -> 16 floats for WMMA N-tile.
__global__ void pad_ws2_kernel(const float* Ws2, const float* bs2, u16* Wp, float* bp) {
  int i = blockIdx.x * blockDim.x + threadIdx.x;
  if (i < 16 * 128) {
    int r = i >> 7, c = i & 127;
    Wp[i] = f2bf(r < 2 ? Ws2[r * 128 + c] : 0.0f);
  }
  if (i < 16) bp[i] = (i < 2) ? bs2[i] : 0.0f;
}

// gl[s][g][0:256] = bf16(gene[g][:]) for all 62 node steps.
__global__ void fill_gl_gene_kernel(const float* gene, u16* gl) {
  int i = blockIdx.x * blockDim.x + threadIdx.x, st = gridDim.x * blockDim.x;
  for (; i < S_STEPS * 256 * 256; i += st) {
    int s = i >> 16, rem = i & 65535, g = rem >> 8, d = rem & 255;
    gl[((size_t)(s * 256 + g)) * 512 + d] = f2bf(gene[rem]);
  }
}

// he ping-pong buf0 <- bf16(h0_edge); ce <- c0_edge (fp32).
__global__ void init_edge_kernel(const float* h0e, const float* c0e, u16* heB, float* ce) {
  int i = blockIdx.x * blockDim.x + threadIdx.x, st = gridDim.x * blockDim.x;
  for (; i < S_STEPS * 131072; i += st) {
    int s = i / 131072, r = i % 131072;
    heB[(size_t)s * 262144 + r] = f2bf(h0e[i]);
    ce[i] = c0e[i];
  }
}

// ---------------------------------------------------------------------------
// Generic WMMA GEMM: out = act(A(MxK,bf16) @ W(NxK,bf16)^T + bias + addsrc)
// One 16x16 tile per wave; 8 waves per 256-thread block.
// ---------------------------------------------------------------------------
__global__ __launch_bounds__(256) void gemm_bf16_kernel(
    const u16* A, int lda, const u16* W, int ldw,
    const float* bias, const float* addsrc, int ldadd,
    float* outF, int ldF, u16* outB, int ldB,
    int M, int N, int K, int act) {
  int wave = threadIdx.x >> 5;
  int tile = blockIdx.x * 8 + wave;
  int ntn = N >> 4;
  if (tile >= (M >> 4) * ntn) return;     // wave-uniform: EXEC stays full
  int m0 = (tile / ntn) << 4;
  int n0 = (tile % ntn) << 4;
  v8f acc = {};
  for (int k = 0; k < K; k += 32)
    acc = WMMA_BF16(load_a_frag(A, lda, m0, k), load_b_frag(W, ldw, n0, k), acc);
  int lane = threadIdx.x & 31;
  int col = n0 + (lane & 15);
  int rb = m0 + ((lane >> 4) << 3);
  float bv = bias ? bias[col] : 0.0f;
#pragma unroll
  for (int r = 0; r < 8; ++r) {
    int row = rb + r;
    float v = acc[r] + bv;
    if (addsrc) v += addsrc[(size_t)row * ldadd + col];
    if (act == 1) v = fmaxf(v, 0.0f);
    if (outF) outF[(size_t)row * ldF + col] = v;
    if (outB) outB[(size_t)row * ldB + col] = f2bf(v);
  }
}

// Node LSTM cell: gates (256x2048 fp32, i|f|g|o) -> c (fp32), h (bf16).
__global__ void lstm_node_update_kernel(const float* gates, float* c, u16* h_bf) {
  int i = blockIdx.x * blockDim.x + threadIdx.x;
  if (i >= 256 * 512) return;
  int g = i >> 9, u = i & 511;
  const float* gr = gates + (size_t)g * 2048;
  float iv = sigm(gr[u]), fv = sigm(gr[512 + u]);
  float gv = tanhf(gr[1024 + u]), ov = sigm(gr[1536 + u]);
  float cn = fv * c[i] + iv * gv;
  c[i] = cn;
  h_bf[i] = f2bf(ov * tanhf(cn));
}

// ---------------------------------------------------------------------------
// In-kernel GEMM phase for the persistent edge-chain kernel.
// ---------------------------------------------------------------------------
__device__ __forceinline__ void gemm_phase(
    const u16* A, int lda, const u16* W, int ldw, const float* bias,
    u16* outB, int ldB, int mt, int nt, int K, bool relu) {
  int wave = threadIdx.x >> 5, lane = threadIdx.x & 31;
  int total = mt * nt;
  for (int t = wave; t < total; t += 32) {
    int m0 = (t / nt) << 4, n0 = (t % nt) << 4;
    v8f acc = {};
    for (int k = 0; k < K; k += 32)
      acc = WMMA_BF16(load_a_frag(A, lda, m0, k), load_b_frag(W, ldw, n0, k), acc);
    int col = n0 + (lane & 15);
    int rb = m0 + ((lane >> 4) << 3);
    float bv = bias[col];
#pragma unroll
    for (int r = 0; r < 8; ++r) {
      float v = acc[r] + bv;
      if (relu) v = fmaxf(v, 0.0f);
      outB[(size_t)(rb + r) * ldB + col] = f2bf(v);
    }
  }
}

#define PHASE_BARRIER() do { __threadfence_block(); __syncthreads(); } while (0)

// ---------------------------------------------------------------------------
// Persistent edge-chain kernel: blockIdx.x = chain s; 32 waves per block.
// Runs j = 0..s sequential edge-LSTM steps; all GEMMs via WMMA bf16.
// ---------------------------------------------------------------------------
__global__ __launch_bounds__(1024, 1) void edge_chain_kernel(
    const u16* Whh_e, const u16* We1, const u16* We2, const u16* We3,
    const u16* Ws1, const u16* Ws2p, const u16* Wf1, const u16* Wf2,
    const float* be1, const float* be2, const float* be3,
    const float* bs1, const float* bs2p, const float* pbf1, const float* pbf2,
    const u16* E_bf,        // (62*256, 2048) bf16: gl@Wih_e^T + biases
    u16* heB,               // (62, 2, 256, 512) bf16 ping-pong
    float* ce,              // (62, 256, 512) fp32
    u16* t1B, u16* t2B, u16* tB, u16* s1B, u16* f1B,
    float* out_ex, float* out_fe) {
  int s = blockIdx.x;
  int wave = threadIdx.x >> 5, lane = threadIdx.x & 31;

  const u16* E = E_bf + (size_t)s * 256 * 2048;
  u16* he0 = heB + (size_t)s * 262144;
  float* ceS = ce + (size_t)s * 131072;
  u16* t1 = t1B + (size_t)s * 131072;
  u16* t2 = t2B + (size_t)s * 65536;
  u16* tt = tB  + (size_t)s * 65536;
  u16* s1 = s1B + (size_t)s * 32768;
  u16* f1 = f1B + (size_t)s * 32768;

  for (int j = 0; j <= s; ++j) {
    const u16* heR = he0 + (j & 1) * 131072;
    u16* heW = he0 + ((j + 1) & 1) * 131072;

    // ---- Phase 1: gates = E + he @ Whh_e^T, fused LSTM cell update. ----
    // 512 units of (16 rows x 16 u-cols); each unit owns all 4 gate tiles.
    for (int ut = wave; ut < 512; ut += 32) {
      int m0 = (ut >> 5) << 4;
      int n0 = (ut & 31) << 4;
      v8f a0 = {}, a1 = {}, a2 = {}, a3 = {};
      for (int k = 0; k < 512; k += 32) {
        v16bf a = load_a_frag(heR, 512, m0, k);
        a0 = WMMA_BF16(a, load_b_frag(Whh_e, 512, n0,        k), a0);
        a1 = WMMA_BF16(a, load_b_frag(Whh_e, 512, 512 + n0,  k), a1);
        a2 = WMMA_BF16(a, load_b_frag(Whh_e, 512, 1024 + n0, k), a2);
        a3 = WMMA_BF16(a, load_b_frag(Whh_e, 512, 1536 + n0, k), a3);
      }
      int col = n0 + (lane & 15);
      int rb = m0 + ((lane >> 4) << 3);
#pragma unroll
      for (int r = 0; r < 8; ++r) {
        int g = rb + r;
        const u16* Eg = E + (size_t)g * 2048;
        float iv = a0[r] + bf2f(Eg[col]);
        float fv = a1[r] + bf2f(Eg[512 + col]);
        float gv = a2[r] + bf2f(Eg[1024 + col]);
        float ov = a3[r] + bf2f(Eg[1536 + col]);
        float cn = sigm(fv) * ceS[g * 512 + col] + sigm(iv) * tanhf(gv);
        ceS[g * 512 + col] = cn;
        heW[g * 512 + col] = f2bf(sigm(ov) * tanhf(cn));
      }
    }
    PHASE_BARRIER();

    // ---- Phase 2: t1 = relu(he @ We1^T + be1)  (256x512, K=512) ----
    gemm_phase(heW, 512, We1, 512, be1, t1, 512, 16, 32, 512, true);
    PHASE_BARRIER();
    // ---- Phase 3: t2 = relu(t1 @ We2^T + be2)  (256x256, K=512) ----
    gemm_phase(t1, 512, We2, 512, be2, t2, 256, 16, 16, 512, true);
    PHASE_BARRIER();
    // ---- Phase 4: t = t2 @ We3^T + be3         (256x256, K=256) ----
    gemm_phase(t2, 256, We3, 256, be3, tt, 256, 16, 16, 256, false);
    PHASE_BARRIER();
    // ---- Phase 5: s1 = relu(t @ Ws1^T + bs1); f1 = relu(t @ Wf1^T + bf1) ----
    gemm_phase(tt, 256, Ws1, 256, bs1, s1, 128, 16, 8, 256, true);
    gemm_phase(tt, 256, Wf1, 256, pbf1, f1, 128, 16, 8, 256, true);
    PHASE_BARRIER();

    // ---- Phase 6: heads + scatter. 128 ft tiles + 16 sc tiles = 144 units ----
    for (int t = wave; t < 144; t += 32) {
      if (t < 128) {  // ft = f1 @ Wf2^T + bf2 -> fe[g, s+2, j, :] / fe[g, j, s+2, :]
        int m0 = (t >> 3) << 4;
        int n0 = (t & 7) << 4;
        v8f acc = {};
        for (int k = 0; k < 128; k += 32)
          acc = WMMA_BF16(load_a_frag(f1, 128, m0, k), load_b_frag(Wf2, 128, n0, k), acc);
        int u = n0 + (lane & 15);
        int d = u & 63;
        int ni = (u >> 6) ? j : s + 2;
        int nj = (u >> 6) ? s + 2 : j;
        int rb = m0 + ((lane >> 4) << 3);
        float bv = pbf2[u];
#pragma unroll
        for (int r = 0; r < 8; ++r) {
          int g = rb + r;
          out_fe[(((size_t)g * 64 + ni) * 64 + nj) * 64 + d] = acc[r] + bv;
        }
      } else {        // sc = sigmoid(s1 @ Ws2p^T + bs2p) -> ex (cols 0,1 valid)
        int m0 = (t - 128) << 4;
        v8f acc = {};
        for (int k = 0; k < 128; k += 32)
          acc = WMMA_BF16(load_a_frag(s1, 128, m0, k), load_b_frag(Ws2p, 128, 0, k), acc);
        int col = lane & 15;
        if (col < 2) {
          int rb = m0 + ((lane >> 4) << 3);
          int ni = col ? j : s + 2;
          int nj = col ? s + 2 : j;
          float bv = bs2p[col];
#pragma unroll
          for (int r = 0; r < 8; ++r) {
            int g = rb + r;
            out_ex[(size_t)g * 4096 + ni * 64 + nj] = sigm(acc[r] + bv);
          }
        }
      }
    }
    PHASE_BARRIER();
  }
}

// ===========================================================================
// Host launcher
// ===========================================================================
extern "C" void kernel_launch(void* const* d_in, const int* in_sizes, int n_in,
                              void* d_out, int out_size, void* d_ws, size_t ws_size,
                              hipStream_t stream) {
  (void)in_sizes; (void)n_in; (void)ws_size;
  const float* in_gene  = (const float*)d_in[0];
  const float* in_Wih_n = (const float*)d_in[1];
  const float* in_Whh_n = (const float*)d_in[2];
  const float* in_bih_n = (const float*)d_in[3];
  const float* in_bhh_n = (const float*)d_in[4];
  const float* in_Wih_e = (const float*)d_in[5];
  const float* in_Whh_e = (const float*)d_in[6];
  const float* in_bih_e = (const float*)d_in[7];
  const float* in_bhh_e = (const float*)d_in[8];
  const float* in_Wn1 = (const float*)d_in[9];
  const float* in_bn1 = (const float*)d_in[10];
  const float* in_Wn2 = (const float*)d_in[11];
  const float* in_bn2 = (const float*)d_in[12];
  const float* in_We1 = (const float*)d_in[13];
  const float* in_be1 = (const float*)d_in[14];
  const float* in_We2 = (const float*)d_in[15];
  const float* in_be2 = (const float*)d_in[16];
  const float* in_We3 = (const float*)d_in[17];
  const float* in_be3 = (const float*)d_in[18];
  const float* in_Ws1 = (const float*)d_in[19];
  const float* in_bs1 = (const float*)d_in[20];
  const float* in_Ws2 = (const float*)d_in[21];
  const float* in_bs2 = (const float*)d_in[22];
  const float* in_Wf1 = (const float*)d_in[23];
  const float* in_bf1 = (const float*)d_in[24];
  const float* in_Wf2 = (const float*)d_in[25];
  const float* in_bf2 = (const float*)d_in[26];
  const float* in_h0n = (const float*)d_in[27];
  const float* in_c0n = (const float*)d_in[28];
  const float* in_h0e = (const float*)d_in[29];
  const float* in_c0e = (const float*)d_in[30];

  float* out = (float*)d_out;
  float* out_ex = out;                 // (256,64,64,1)
  float* out_fe = out + 1048576;       // (256,64,64,64)

  // ---- Workspace carve (256B aligned) ----
  char* base = (char*)d_ws;
  size_t off = 0;
  auto carve = [&](size_t bytes) -> char* {
    char* p = base + off;
    off += (bytes + 255) & ~(size_t)255;
    return p;
  };
  // fp32
  float* Xn     = (float*)carve(256 * 2048 * 4);      // gene@Wih_n^T + biases
  float* gatesN = (float*)carve(256 * 2048 * 4);
  float* cN     = (float*)carve(256 * 512 * 4);
  float* bnF    = (float*)carve(2048 * 4);
  float* beF    = (float*)carve(2048 * 4);
  float* bs2p   = (float*)carve(16 * 4);
  float* ce     = (float*)carve((size_t)S_STEPS * 131072 * 4);
  // bf16 weights
  u16* wihn_b = (u16*)carve(2048 * 256 * 2);
  u16* whhn_b = (u16*)carve(2048 * 512 * 2);
  u16* wn1_b  = (u16*)carve(256 * 512 * 2);
  u16* wn2_b  = (u16*)carve(256 * 256 * 2);
  u16* wihe_b = (u16*)carve(2048 * 512 * 2);
  u16* whhe_b = (u16*)carve(2048 * 512 * 2);
  u16* we1_b  = (u16*)carve(512 * 512 * 2);
  u16* we2_b  = (u16*)carve(256 * 512 * 2);
  u16* we3_b  = (u16*)carve(256 * 256 * 2);
  u16* ws1_b  = (u16*)carve(128 * 256 * 2);
  u16* ws2p_b = (u16*)carve(16 * 128 * 2);
  u16* wf1_b  = (u16*)carve(128 * 256 * 2);
  u16* wf2_b  = (u16*)carve(128 * 128 * 2);
  // bf16 activations
  u16* gene_b = (u16*)carve(256 * 256 * 2);
  u16* hN_b   = (u16*)carve(256 * 512 * 2);
  u16* e1_b   = (u16*)carve(256 * 256 * 2);
  u16* gl_b   = (u16*)carve((size_t)S_STEPS * 131072 * 2);   // [gene | entry]
  u16* E_b    = (u16*)carve((size_t)S_STEPS * 256 * 2048 * 2);
  u16* heB    = (u16*)carve((size_t)S_STEPS * 262144 * 2);   // ping-pong
  u16* t1_b   = (u16*)carve((size_t)S_STEPS * 131072 * 2);
  u16* t2_b   = (u16*)carve((size_t)S_STEPS * 65536 * 2);
  u16* t_b    = (u16*)carve((size_t)S_STEPS * 65536 * 2);
  u16* s1_b   = (u16*)carve((size_t)S_STEPS * 32768 * 2);
  u16* f1_b   = (u16*)carve((size_t)S_STEPS * 32768 * 2);

  auto cvt = [&](const float* s, u16* d, int n) {
    int blk = (n + 255) / 256; if (blk > 4096) blk = 4096;
    cvt_bf16_kernel<<<blk, 256, 0, stream>>>(s, d, n);
  };
  auto gemm = [&](const u16* A, int lda, const u16* W, int ldw, const float* bias,
                  const float* addsrc, int ldadd, float* outF, int ldF,
                  u16* outB, int ldB, int M, int N, int K, int act) {
    int tiles = (M >> 4) * (N >> 4);
    gemm_bf16_kernel<<<(tiles + 7) >> 3, 256, 0, stream>>>(
        A, lda, W, ldw, bias, addsrc, ldadd, outF, ldF, outB, ldB, M, N, K, act);
  };

  // ---- 0. zero output (harness poisons it) ----
  zero_f32_kernel<<<4096, 256, 0, stream>>>(out, (long long)out_size);

  // ---- 1. prep: weights -> bf16, fused biases, padded Ws2, init state ----
  cvt(in_Wih_n, wihn_b, 2048 * 256);
  cvt(in_Whh_n, whhn_b, 2048 * 512);
  cvt(in_Wn1, wn1_b, 256 * 512);
  cvt(in_Wn2, wn2_b, 256 * 256);
  cvt(in_Wih_e, wihe_b, 2048 * 512);
  cvt(in_Whh_e, whhe_b, 2048 * 512);
  cvt(in_We1, we1_b, 512 * 512);
  cvt(in_We2, we2_b, 256 * 512);
  cvt(in_We3, we3_b, 256 * 256);
  cvt(in_Ws1, ws1_b, 128 * 256);
  cvt(in_Wf1, wf1_b, 128 * 256);
  cvt(in_Wf2, wf2_b, 128 * 128);
  cvt(in_gene, gene_b, 256 * 256);
  cvt(in_h0n, hN_b, 256 * 512);
  add_f32_kernel<<<8, 256, 0, stream>>>(in_bih_n, in_bhh_n, bnF, 2048);
  add_f32_kernel<<<8, 256, 0, stream>>>(in_bih_e, in_bhh_e, beF, 2048);
  pad_ws2_kernel<<<8, 256, 0, stream>>>(in_Ws2, in_bs2, ws2p_b, bs2p);
  fill_gl_gene_kernel<<<4096, 256, 0, stream>>>(in_gene, gl_b);
  copy_f32_kernel<<<512, 256, 0, stream>>>(in_c0n, cN, 256 * 512);
  init_edge_kernel<<<4096, 256, 0, stream>>>(in_h0e, in_c0e, heB, ce);

  // ---- 2. Xn = gene @ Wih_n^T + (bih_n + bhh_n): invariant over node steps ----
  gemm(gene_b, 256, wihn_b, 256, bnF, nullptr, 0, Xn, 2048, nullptr, 0,
       256, 2048, 256, 0);

  // ---- 3. Node LSTM chain (sequential, cheap): 62 steps ----
  for (int s = 0; s < S_STEPS; ++s) {
    // gates = Xn + h @ Whh_n^T
    gemm(hN_b, 512, whhn_b, 512, nullptr, Xn, 2048, gatesN, 2048, nullptr, 0,
         256, 2048, 512, 0);
    lstm_node_update_kernel<<<512, 256, 0, stream>>>(gatesN, cN, hN_b);
    // entry = relu(h @ Wn1^T + bn1) @ Wn2^T + bn2 -> gl[s][:, 256:512]
    gemm(hN_b, 512, wn1_b, 512, in_bn1, nullptr, 0, nullptr, 0, e1_b, 256,
         256, 256, 512, 1);
    gemm(e1_b, 256, wn2_b, 256, in_bn2, nullptr, 0, nullptr, 0,
         gl_b + (size_t)s * 131072 + 256, 512, 256, 256, 256, 0);
  }

  // ---- 4. E = gl @ Wih_e^T + (bih_e + bhh_e): one batched GEMM over all s ----
  gemm(gl_b, 512, wihe_b, 512, beF, nullptr, 0, nullptr, 0, E_b, 2048,
       S_STEPS * 256, 2048, 512, 0);

  // ---- 5. 62 independent edge-LSTM chains, one persistent WG each ----
  edge_chain_kernel<<<S_STEPS, 1024, 0, stream>>>(
      whhe_b, we1_b, we2_b, we3_b, ws1_b, ws2p_b, wf1_b, wf2_b,
      in_be1, in_be2, in_be3, in_bs1, bs2p, in_bf1, in_bf2,
      E_b, heB, ce, t1_b, t2_b, t_b, s1_b, f1_b, out_ex, out_fe);
}
